// SelfAttention_63677185130723
// MI455X (gfx1250) — matrix operand order repaired and checked by reading, not compiled, not run on hardware
//
#include <hip/hip_runtime.h>

typedef __attribute__((ext_vector_type(16))) __bf16 v16bf;
typedef __attribute__((ext_vector_type(8)))  float  v8f;

#define NB 8
#define NC 128
#define NH 4
#define HD 32
#define NS 2304  // 48*48

struct U128 { unsigned int x, y, z, w; };
union BF16x16 { v16bf v; __bf16 e[16]; U128 q[2]; };

// ---------------------------------------------------------------------------
// Phase 0: one-shot f32 -> bf16 conversion of the four 128x128 weight mats.
// Row-major layout is exactly what the WMMA B-tile loads want (16 contiguous
// bf16 per lane). 64K elements total; trivially memory-bound, runs once.
// ---------------------------------------------------------------------------
__global__ __launch_bounds__(256) void cvt_weights_kernel(
    const float* __restrict__ Wq, const float* __restrict__ Wk,
    const float* __restrict__ Wv, const float* __restrict__ Wo,
    __bf16* __restrict__ wqb, __bf16* __restrict__ wkb,
    __bf16* __restrict__ wvb, __bf16* __restrict__ wob)
{
  int i = blockIdx.x * 256 + threadIdx.x;   // 0 .. 16383
  wqb[i] = (__bf16)Wq[i];
  wkb[i] = (__bf16)Wk[i];
  wvb[i] = (__bf16)Wv[i];
  wob[i] = (__bf16)Wo[i];
}

// ---------------------------------------------------------------------------
// Phase 1: q/k/v = xf @ W^T + b   (xf[b,s,c] = x[b,c,s]), bf16 outputs.
// q,k stored [b,h,s,d]; v stored transposed [b,h,d,s] for phase-2 B-tiles.
// ---------------------------------------------------------------------------
__global__ __launch_bounds__(32) void qkv_proj_kernel(
    const float* __restrict__ x,
    const __bf16* __restrict__ wqb, const float* __restrict__ bq,
    const __bf16* __restrict__ wkb, const float* __restrict__ bk,
    const __bf16* __restrict__ wvb, const float* __restrict__ bv,
    __bf16* __restrict__ qb, __bf16* __restrict__ kb,
    __bf16* __restrict__ vtb)
{
  const int lane = threadIdx.x;
  const int half = lane >> 4;   // 0: lanes 0-15, 1: lanes 16-31
  const int l16  = lane & 15;
  const int s0   = blockIdx.x * 16;
  const int b    = blockIdx.y;

  // A tiles (16 s-rows x 32 c_in), 4 K-steps. A layout: lane row = l16,
  // element e -> K = e + 16*half  (contiguous order, matches B).
  BF16x16 a[4];
  const float* xb = x + (size_t)b * NC * NS;
#pragma unroll
  for (int kc = 0; kc < 4; ++kc) {
#pragma unroll
    for (int e = 0; e < 16; ++e) {
      int kin = kc * 32 + e + 16 * half;
      a[kc].e[e] = (__bf16)xb[(size_t)kin * NS + s0 + l16];
    }
  }

  const __bf16* Ws[3] = { wqb, wkb, wvb };
  const float*  bs[3] = { bq, bk, bv };

#pragma unroll
  for (int w = 0; w < 3; ++w) {
    const __bf16* W  = Ws[w];
    const float*  bi = bs[w];
    for (int n0 = 0; n0 < 8; ++n0) {
      v8f acc = {};
#pragma unroll
      for (int kc = 0; kc < 4; ++kc) {
        // B layout (32x16 bf16): element e -> K = e + 16*half, N = l16.
        // B[kk,n] = W[n0*16+n, kc*32+kk] -> 16 contiguous bf16 per lane.
        BF16x16 bm;
        const __bf16* wrow = W + (size_t)(n0 * 16 + l16) * NC + kc * 32 + 16 * half;
        bm.q[0] = *(const U128*)(wrow);
        bm.q[1] = *(const U128*)(wrow + 8);
        acc = __builtin_amdgcn_wmma_f32_16x16x32_bf16(
            false, a[kc].v, false, bm.v, (short)0, acc, false, false);
      }
      const int   c    = n0 * 16 + l16;         // output channel for all 8 rows
      const float bias = bi[c];
      const int   hh   = c >> 5, d = c & 31;
#pragma unroll
      for (int i = 0; i < 8; ++i) {             // D row = i + 8*half
        int s = s0 + i + 8 * half;
        __bf16 val = (__bf16)(acc[i] + bias);
        if (w == 0)      qb [(((size_t)b * NH + hh) * NS + s) * HD + d] = val;
        else if (w == 1) kb [(((size_t)b * NH + hh) * NS + s) * HD + d] = val;
        else             vtb[(((size_t)b * NH + hh) * HD + d) * NS + s] = val;
      }
    }
  }
}

// ---------------------------------------------------------------------------
// Phase 2: flash attention. One wave = one (b, h, 16-query tile).
// hd == 32 -> one bf16 WMMA per 16x16 score tile. Streamed softmax; the
// 680 MB score matrix never exists in memory.
// ---------------------------------------------------------------------------
__global__ __launch_bounds__(32) void flash_attn_kernel(
    const __bf16* __restrict__ qb,
    const __bf16* __restrict__ kb,
    const __bf16* __restrict__ vtb,
    __bf16* __restrict__ ctxb)
{
  __shared__ __bf16 plds[16 * 32];   // P-tile repack buffer (1 KB)

  const int lane = threadIdx.x;
  const int half = lane >> 4;
  const int l16  = lane & 15;
  const int q0   = blockIdx.x * 16;
  const int h    = blockIdx.y;
  const int b    = blockIdx.z;
  const size_t bh = (size_t)b * NH + h;

  // Q as A-matrix: row = l16, element e -> d = e + 16*half
  BF16x16 aq;
  const __bf16* qrow = qb + (bh * NS + q0 + l16) * HD;
  aq.q[0] = *(const U128*)(qrow + 16 * half);
  aq.q[1] = *(const U128*)(qrow + 16 * half + 8);

  const float scale = 0.17677669529663688f;  // 1/sqrt(32)

  v8f ctx0 = {}, ctx1 = {};
  float m[8], l[8];
#pragma unroll
  for (int i = 0; i < 8; ++i) { m[i] = -1e30f; l[i] = 0.0f; }

  const __bf16* kbase = kb  + bh * NS * HD;
  const __bf16* vbase = vtb + bh * HD * NS;

  for (int j = 0; j < NS; j += 32) {
    // K^T as B-matrix: element e -> K(d) = e + 16*half, N(key) = l16.
    BF16x16 bk0, bk1;
    const __bf16* kr0 = kbase + (size_t)(j + l16) * HD + 16 * half;
    const __bf16* kr1 = kbase + (size_t)(j + 16 + l16) * HD + 16 * half;
    bk0.q[0] = *(const U128*)(kr0);
    bk0.q[1] = *(const U128*)(kr0 + 8);
    bk1.q[0] = *(const U128*)(kr1);
    bk1.q[1] = *(const U128*)(kr1 + 8);

    v8f z = {};
    v8f s0t = __builtin_amdgcn_wmma_f32_16x16x32_bf16(
        false, aq.v, false, bk0.v, (short)0, z, false, false);
    v8f s1t = __builtin_amdgcn_wmma_f32_16x16x32_bf16(
        false, aq.v, false, bk1.v, (short)0, z, false, false);

    // Online softmax. Accumulator element (i, lane) belongs to row i+8*half;
    // row reductions = 16-lane xor reductions (stay inside each half).
    float p0[8], p1[8];
#pragma unroll
    for (int i = 0; i < 8; ++i) {
      float a0 = s0t[i] * scale, a1 = s1t[i] * scale;
      float mx = fmaxf(a0, a1);
      mx = fmaxf(mx, __shfl_xor(mx, 1, 32));
      mx = fmaxf(mx, __shfl_xor(mx, 2, 32));
      mx = fmaxf(mx, __shfl_xor(mx, 4, 32));
      mx = fmaxf(mx, __shfl_xor(mx, 8, 32));
      float mn    = fmaxf(m[i], mx);
      float alpha = __expf(m[i] - mn);
      p0[i] = __expf(a0 - mn);
      p1[i] = __expf(a1 - mn);
      float r = p0[i] + p1[i];
      r += __shfl_xor(r, 1, 32);
      r += __shfl_xor(r, 2, 32);
      r += __shfl_xor(r, 4, 32);
      r += __shfl_xor(r, 8, 32);
      l[i] = l[i] * alpha + r;
      m[i] = mn;
      ctx0[i] *= alpha;
      ctx1[i] *= alpha;
    }

    // Repack P (C/D layout) -> A layout through LDS.
    __syncthreads();
#pragma unroll
    for (int i = 0; i < 8; ++i) {
      int row = i + 8 * half;
      plds[row * 32 + l16]      = (__bf16)p0[i];   // key cols j..j+15
      plds[row * 32 + 16 + l16] = (__bf16)p1[i];   // key cols j+16..j+31
    }
    __syncthreads();

    BF16x16 ap;   // same A mapping as the q load, 16B-aligned LDS reads
    ap.q[0] = *(const U128*)(&plds[l16 * 32 + 16 * half]);
    ap.q[1] = *(const U128*)(&plds[l16 * 32 + 16 * half + 8]);

    // V as B-matrix: element e -> K(key) = j + e + 16*half, N(d) = l16.
    BF16x16 bv0, bv1;
    const __bf16* vr0 = vbase + (size_t)l16 * NS + j + 16 * half;
    const __bf16* vr1 = vbase + (size_t)(16 + l16) * NS + j + 16 * half;
    bv0.q[0] = *(const U128*)(vr0);
    bv0.q[1] = *(const U128*)(vr0 + 8);
    bv1.q[0] = *(const U128*)(vr1);
    bv1.q[1] = *(const U128*)(vr1 + 8);

    ctx0 = __builtin_amdgcn_wmma_f32_16x16x32_bf16(
        false, ap.v, false, bv0.v, (short)0, ctx0, false, false);
    ctx1 = __builtin_amdgcn_wmma_f32_16x16x32_bf16(
        false, ap.v, false, bv1.v, (short)0, ctx1, false, false);
  }

  // Normalize and store merged ctx[b, s, c] (c = h*32 + d) in bf16.
#pragma unroll
  for (int i = 0; i < 8; ++i) {
    float inv = 1.0f / l[i];
    int s = q0 + i + 8 * half;
    size_t base = ((size_t)b * NS + s) * NC + h * HD;
    ctxb[base + l16]      = (__bf16)(ctx0[i] * inv);
    ctxb[base + 16 + l16] = (__bf16)(ctx1[i] * inv);
  }
}

// ---------------------------------------------------------------------------
// Phase 3: out = ctx @ Wo^T + bo, stored back as [B, C, H, W] fp32.
// ---------------------------------------------------------------------------
__global__ __launch_bounds__(32) void out_proj_kernel(
    const __bf16* __restrict__ ctxb,
    const __bf16* __restrict__ wob, const float* __restrict__ bo,
    float* __restrict__ out)
{
  const int lane = threadIdx.x;
  const int half = lane >> 4;
  const int l16  = lane & 15;
  const int s0   = blockIdx.x * 16;
  const int b    = blockIdx.y;

  BF16x16 a[4];
  const __bf16* crow = ctxb + ((size_t)b * NS + s0 + l16) * NC;
#pragma unroll
  for (int kc = 0; kc < 4; ++kc) {
    a[kc].q[0] = *(const U128*)(crow + kc * 32 + 16 * half);
    a[kc].q[1] = *(const U128*)(crow + kc * 32 + 16 * half + 8);
  }

  for (int n0 = 0; n0 < 8; ++n0) {
    v8f acc = {};
#pragma unroll
    for (int kc = 0; kc < 4; ++kc) {
      BF16x16 bm;
      const __bf16* wrow = wob + (size_t)(n0 * 16 + l16) * NC + kc * 32 + 16 * half;
      bm.q[0] = *(const U128*)(wrow);
      bm.q[1] = *(const U128*)(wrow + 8);
      acc = __builtin_amdgcn_wmma_f32_16x16x32_bf16(
          false, a[kc].v, false, bm.v, (short)0, acc, false, false);
    }
    const int   c    = n0 * 16 + l16;
    const float bias = bo[c];
#pragma unroll
    for (int i = 0; i < 8; ++i) {
      int s = s0 + i + 8 * half;                 // contiguous in s per lane
      out[((size_t)b * NC + c) * NS + s] = acc[i] + bias;
    }
  }
}

// ---------------------------------------------------------------------------
extern "C" void kernel_launch(void* const* d_in, const int* in_sizes, int n_in,
                              void* d_out, int out_size, void* d_ws, size_t ws_size,
                              hipStream_t stream) {
  const float* x  = (const float*)d_in[0];
  const float* Wq = (const float*)d_in[1];
  const float* bq = (const float*)d_in[2];
  const float* Wk = (const float*)d_in[3];
  const float* bk = (const float*)d_in[4];
  const float* Wv = (const float*)d_in[5];
  const float* bv = (const float*)d_in[6];
  const float* Wo = (const float*)d_in[7];
  const float* bo = (const float*)d_in[8];
  float* out = (float*)d_out;

  // Workspace layout (bf16 elements):
  //   4 weight mats of 128*128          =  65,536 elems (128 KB)
  //   q, k, vT, ctx: B*nh*S*hd each     = 2,359,296 elems each (4.5 MB each)
  const size_t NW = (size_t)NC * NC;            // 16384
  const size_t NE = (size_t)NB * NH * NS * HD;  // 2359296
  __bf16* wqb  = (__bf16*)d_ws;
  __bf16* wkb  = wqb + NW;
  __bf16* wvb  = wkb + NW;
  __bf16* wob  = wvb + NW;
  __bf16* qb   = wob + NW;
  __bf16* kb   = qb  + NE;
  __bf16* vtb  = kb  + NE;
  __bf16* ctxb = vtb + NE;

  cvt_weights_kernel<<<dim3(NC * NC / 256), 256, 0, stream>>>(
      Wq, Wk, Wv, Wo, wqb, wkb, wvb, wob);
  qkv_proj_kernel<<<dim3(NS / 16, NB), 32, 0, stream>>>(
      x, wqb, bq, wkb, bk, wvb, bv, qb, kb, vtb);
  flash_attn_kernel<<<dim3(NS / 16, NH, NB), 32, 0, stream>>>(
      qb, kb, vtb, ctxb);
  out_proj_kernel<<<dim3(NS / 16, NB), 32, 0, stream>>>(
      ctxb, Wo ? wob : wob, bo, out);
}
